// MYLSTM_1_29489245454889
// MI455X (gfx1250) — matrix operand order repaired and checked by reading, not compiled, not run on hardware
//
#include <hip/hip_runtime.h>
#include <hip/hip_bf16.h>

// ---------------------------------------------------------------------------
// B-spline-parameterized LSTM for MI455X (gfx1250, wave32, WMMA).
//
//  per step t:
//    W_t[g,d,u] = sum_q bt[t,q] * Q_g[d, u*8+q]           (tiny contraction)
//    g[g,b,u]   = [S_prev | x_t](b,d) @ W_t[g,d,u]         (256 x 2048 x 1024)
//    C = sig(g0)*S_prev + sig(g1)*sig(g2); S=tanh(C); H = tanh(g3)*S
//
//  - per-step bf16 weight precombine (4 MB ws) kills the 8x q-redundancy
//  - bf16 WMMA 16x16x32 w/ fp32 accum; wave owns all 4 gates of a 16x16 tile
//  - W tiles staged to LDS via global_load_async_to_lds_b128 (ASYNCcnt)
//  - activations via hardware v_tanh_f32 (CDNA5 TRANS op)
// ---------------------------------------------------------------------------

typedef __bf16 bf16;
typedef __attribute__((ext_vector_type(16))) __bf16 v16bf;
typedef __attribute__((ext_vector_type(8)))  __bf16 v8bf;
typedef __attribute__((ext_vector_type(8)))  float  v8f;

#define T_STEPS 40
#define QB      8
#define UNITS   512
#define FEAT    512
#define BATCH   256
#define DTOT    (UNITS + FEAT)   // 1024

// CDNA5 hardware tanh. TRANS-op hazard rule (ISA 7.4): one independent op /
// V_NOP before the result may be consumed -> embed the v_nop in the asm.
__device__ __forceinline__ float hw_tanh(float x) {
    float r;
    asm("v_tanh_f32 %0, %1\n\tv_nop" : "=v"(r) : "v"(x));
    return r;
}
__device__ __forceinline__ float hw_sigmoid(float x) {
    return 0.5f + 0.5f * hw_tanh(0.5f * x);
}

// ---------------------------------------------------------------------------
// Kernel 1: combine B-spline basis into per-step bf16 weights.
//   Wc[g][u][k] (k contiguous) <- sum_q bt[q] * Qg[k][u*8+q]
// k-contiguous layout matches the WMMA B-fragment (one 32B ds_load per lane).
// ---------------------------------------------------------------------------
__global__ __launch_bounds__(256) void bspline_combine(
    const float* __restrict__ Q1, const float* __restrict__ Q2,
    const float* __restrict__ Q3, const float* __restrict__ Q4,
    const float* __restrict__ Bt, int t, bf16* __restrict__ Wc)
{
    __shared__ bf16 tile[64 * 65];          // [u][k] transpose staging (+pad)
    const int g   = blockIdx.z;
    const float* Q = (g == 0) ? Q1 : (g == 1) ? Q2 : (g == 2) ? Q3 : Q4;
    const int dd0 = blockIdx.x * 64;        // k tile  (DTOT/64 = 16)
    const int u0  = blockIdx.y * 64;        // u tile  (UNITS/64 = 8)

    float b[QB];
#pragma unroll
    for (int q = 0; q < QB; ++q) b[q] = Bt[t * QB + q];   // uniform s_loads

    for (int i = threadIdx.x; i < 64 * 64; i += 256) {
        const int ldd = i >> 6, lu = i & 63;
        const float* src = Q + (size_t)(dd0 + ldd) * (UNITS * QB)
                             + (size_t)(u0 + lu) * QB;    // coalesced 32B reads
        float s = 0.f;
#pragma unroll
        for (int q = 0; q < QB; ++q) s += b[q] * src[q];
        tile[lu * 65 + ldd] = (bf16)s;
    }
    __syncthreads();
    for (int i = threadIdx.x; i < 64 * 64; i += 256) {
        const int lu = i >> 6, ldd = i & 63;
        Wc[((size_t)g * UNITS + (u0 + lu)) * DTOT + dd0 + ldd] = tile[lu * 65 + ldd];
    }
}

// ---------------------------------------------------------------------------
// Kernel 2: one recurrence step.
//   grid = (BATCH/16, UNITS/64), block = 128 (4 waves)
// ---------------------------------------------------------------------------
__global__ __launch_bounds__(128) void lstm_step(
    const float* __restrict__ x,      // (B, T, F)
    const bf16*  __restrict__ Wc,     // [4][UNITS][DTOT] bf16
    const float* __restrict__ Sprev,  // (B, U)  state = tanh(C)
    float*       __restrict__ Snext,  // (B, U)
    float*       __restrict__ Hout,   // (B, T, U)
    int t)
{
    __shared__ bf16 ldsA[16 * 64];        // [m][k]    2 KB
    __shared__ bf16 ldsW[4 * 64 * 64];    // [g][u][k] 32 KB

    const int m0   = blockIdx.x * 16;     // batch tile
    const int n0   = blockIdx.y * 64;     // unit tile
    const int tid  = threadIdx.x;
    const int lane = tid & 31;
    const int w    = tid >> 5;            // wave 0..3 -> u sub-tile

    // LDS byte base of ldsW (flat LDS aperture: low 32 bits == LDS offset)
    const unsigned ldsW_base = (unsigned)(uintptr_t)(void*)&ldsW[0];

    v8f acc[4] = {v8f{}, v8f{}, v8f{}, v8f{}};

    for (int k0 = 0; k0 < DTOT; k0 += 64) {
        __syncthreads();
        // ---- stage A = [S_prev | x_t] rows (fp32 -> bf16, manual path) ----
        for (int i = tid; i < 16 * 64; i += 128) {
            const int m = i >> 6, kk = i & 63;
            const int k = k0 + kk;
            const int bb = m0 + m;
            const float v = (k < UNITS)
                ? Sprev[(size_t)bb * UNITS + k]
                : x[((size_t)bb * T_STEPS + t) * FEAT + (k - UNITS)];
            ldsA[m * 64 + kk] = (bf16)v;
        }
        // ---- stage W tiles via async global->LDS B128 copies (ASYNCcnt) ----
        for (int i = tid; i < 4 * 64 * 8; i += 128) {
            const int g   = i >> 9;            // /(64*8)
            const int rem = i & 511;
            const int ul  = rem >> 3;
            const int kv  = rem & 7;
            const bf16* src = Wc + ((size_t)g * UNITS + n0 + ul) * DTOT
                                 + k0 + kv * 8;
            const unsigned dst = ldsW_base
                + (unsigned)(((g * 64 + ul) * 64 + kv * 8) * (int)sizeof(bf16));
            asm volatile("global_load_async_to_lds_b128 %0, %1, off"
                         :: "v"(dst), "v"(src) : "memory");
        }
        asm volatile("s_wait_asynccnt 0" ::: "memory");
        __syncthreads();

        const int mrow  = lane & 15;
        const int khalf = lane >> 4;       // contiguous k: lo lanes K0-15, hi K16-31
#pragma unroll
        for (int c = 0; c < 2; ++c) {      // two 16x16x32 WMMAs cover K=64
            union { v16bf v; v8bf h[2]; } a;     // A-fragment, contiguous k = 16*khalf + s
            a.h[0] = *(const v8bf*)(&ldsA[mrow * 64 + c * 32 + khalf * 16]);
            a.h[1] = *(const v8bf*)(&ldsA[mrow * 64 + c * 32 + khalf * 16 + 8]);
#pragma unroll
            for (int g = 0; g < 4; ++g) {
                const int ul = w * 16 + (lane & 15);   // column owned by lane
                const v16bf bfrag = *(const v16bf*)(
                    &ldsW[(g * 64 + ul) * 64 + c * 32 + khalf * 16]);
                acc[g] = __builtin_amdgcn_wmma_f32_16x16x32_bf16(
                    false, a.v, false, bfrag, (short)0, acc[g], false, false);
            }
        }
    }

    // ---- cell update; C/D layout: vgpr r, lane l -> m=r+8*(l/16), n=l%16 ----
    const int nloc = lane & 15;
    const int mhi  = (lane >> 4) * 8;
    const int u    = n0 + w * 16 + nloc;
#pragma unroll
    for (int r = 0; r < 8; ++r) {
        const int bb = m0 + mhi + r;
        const float f  = hw_sigmoid(acc[0][r]);
        const float ig = hw_sigmoid(acc[1][r]);
        const float gg = hw_sigmoid(acc[2][r]);
        const float o  = hw_tanh(acc[3][r]);
        const float sp = Sprev[(size_t)bb * UNITS + u];
        const float c  = ig * gg + f * sp;
        const float sn = hw_tanh(c);
        Snext[(size_t)bb * UNITS + u]                = sn;
        Hout[((size_t)bb * T_STEPS + t) * UNITS + u] = o * sn;
    }
}

// ---------------------------------------------------------------------------
extern "C" void kernel_launch(void* const* d_in, const int* in_sizes, int n_in,
                              void* d_out, int out_size, void* d_ws, size_t ws_size,
                              hipStream_t stream) {
    const float* x  = (const float*)d_in[0];   // (256, 40, 512)
    const float* Q1 = (const float*)d_in[1];   // (1024, 4096)
    const float* Q2 = (const float*)d_in[2];
    const float* Q3 = (const float*)d_in[3];
    const float* Q4 = (const float*)d_in[4];
    const float* Bt = (const float*)d_in[5];   // (40, 8)
    float* out = (float*)d_out;                // (256, 40, 512) fp32
    (void)out;

    // workspace: Wc (4 MiB bf16) + two fp32 state buffers (512 KiB each)
    bf16*  Wc = (bf16*)d_ws;
    float* Sa = (float*)((char*)d_ws + (size_t)4 * UNITS * DTOT * sizeof(bf16));
    float* Sb = Sa + (size_t)BATCH * UNITS;

    hipMemsetAsync(Sa, 0, (size_t)BATCH * UNITS * sizeof(float), stream);

    const dim3 cgrid(DTOT / 64, UNITS / 64, 4);   // 16 x 8 x 4
    const dim3 sgrid(BATCH / 16, UNITS / 64);     // 16 x 8

    for (int t = 0; t < T_STEPS; ++t) {
        bspline_combine<<<cgrid, 256, 0, stream>>>(Q1, Q2, Q3, Q4, Bt, t, Wc);
        const float* Sp = (t & 1) ? Sb : Sa;
        float*       Sn = (t & 1) ? Sa : Sb;
        lstm_step<<<sgrid, 128, 0, stream>>>(x, Wc, Sp, Sn, (float*)d_out, t);
    }
}
